// VectorQuantizer_34926674051808
// MI455X (gfx1250) — matrix operand order repaired and checked by reading, not compiled, not run on hardware
//
#include <hip/hip_runtime.h>
#include <hip/hip_bf16.h>
#include <stdint.h>

typedef __attribute__((ext_vector_type(16))) _Float16 v16h;
typedef __attribute__((ext_vector_type(8)))  float    v8f;
typedef __attribute__((ext_vector_type(4)))  float    v4f;
typedef __attribute__((ext_vector_type(4)))  int      v4i;

#define VQ_N 65536
#define VQ_K 8192
#define VQ_D 256
#define NTILE (VQ_N / 16)     // 4096 row tiles of z
#define KTILE (VQ_K / 16)     // 512 codeword tiles
#define CH    (VQ_D / 32)     // 8 K-chunks of 32 along D
#define STAGE 4               // ktiles staged in LDS per buffer (4 * 8KB = 32KB)
#define NSTAGE (KTILE / STAGE)

// ---------------- async global->LDS copy (guarded) ----------------
#if defined(__HIP_DEVICE_COMPILE__) && __has_builtin(__builtin_amdgcn_global_load_async_to_lds_b128)
#define VQ_HAVE_ASYNC 1
#define ASYNC_COPY16(dst, src)                                                  \
  __builtin_amdgcn_global_load_async_to_lds_b128(                               \
      (__attribute__((address_space(1))) v4i*)(uintptr_t)(src),                 \
      (__attribute__((address_space(3))) v4i*)(uint32_t)(uintptr_t)(dst), 0, 0)
#else
#define VQ_HAVE_ASYNC 0
#define ASYNC_COPY16(dst, src) *(v4i*)(dst) = *(const v4i*)(src)
#endif

__device__ __forceinline__ void vq_wait_async() {
#if defined(__HIP_DEVICE_COMPILE__)
#if VQ_HAVE_ASYNC
#if __has_builtin(__builtin_amdgcn_s_wait_asynccnt)
  __builtin_amdgcn_s_wait_asynccnt(0);
#else
  asm volatile("s_wait_asynccnt 0" ::: "memory");
#endif
#endif
#endif
}

// ---------------- pack z (f32 -> f16, WMMA A-tile layout) ----------------
// A 16x32 f16 tile layout (per ISA): lanes 0-15 hold rows M=lane, K pattern
// v<4 -> K=(v*2+h), v>=4 -> K=16+((v-4)*2+h); lanes 16-31 same rows, K+8.
__global__ __launch_bounds__(256) void vq_pack_z(const float* __restrict__ z,
                                                 _Float16* __restrict__ pa) {
  int t    = blockIdx.x * 256 + threadIdx.x;   // one thread per (tile,chunk,lane)
  int lane = t & 31;
  int tc   = t >> 5;
  int c    = tc & (CH - 1);
  int r    = tc / CH;
  const float* src = z + (size_t)(r * 16 + (lane & 15)) * VQ_D + c * 32;
  int kbase = (lane >= 16) ? 8 : 0;
  _Float16 vals[16];
#pragma unroll
  for (int v = 0; v < 8; ++v)
#pragma unroll
    for (int h = 0; h < 2; ++h) {
      int K = ((v >= 4) ? 16 : 0) + kbase + (v & 3) * 2 + h;
      vals[v * 2 + h] = (_Float16)src[K];
    }
  *(v16h*)(pa + (size_t)t * 16) = *(v16h*)vals;
}

// ---------------- pack emb (f32 -> f16, WMMA B-tile layout) ----------------
// B 32x16 f16: lane l holds column (l&15), K = (l>>4)*8 + j for j<8, 16 + (l>>4)*8 + (j-8) for j>=8.
__global__ __launch_bounds__(256) void vq_pack_emb(const float* __restrict__ emb,
                                                   _Float16* __restrict__ pb) {
  int t    = blockIdx.x * 256 + threadIdx.x;
  int lane = t & 31;
  int tc   = t >> 5;
  int c    = tc & (CH - 1);
  int kt   = tc / CH;
  int col  = kt * 16 + (lane & 15);
  int dbase = c * 32 + (lane >> 4) * 8;
  const float* src = emb + (size_t)col * VQ_D + dbase;
  _Float16 vals[16];
#pragma unroll
  for (int j = 0; j < 16; ++j) vals[j] = (_Float16)src[j < 8 ? j : j + 8];
  *(v16h*)(pb + (size_t)t * 16) = *(v16h*)vals;
}

// ---------------- e2[k] = ||emb_k||^2 ----------------
__global__ __launch_bounds__(256) void vq_rownorm(const float* __restrict__ emb,
                                                  float* __restrict__ e2) {
  int k = blockIdx.x * 256 + threadIdx.x;
  const v4f* row = (const v4f*)(emb + (size_t)k * VQ_D);
  float s = 0.f;
#pragma unroll 4
  for (int i = 0; i < VQ_D / 4; ++i) {
    v4f v = row[i];
    s += v.x * v.x + v.y * v.y + v.z * v.z + v.w * v.w;
  }
  e2[k] = s;
}

// ---------------- fused WMMA GEMM + argmin ----------------
__device__ __forceinline__ void vq_fill_stage(const _Float16* __restrict__ pb,
                                              _Float16* dst, int s, int tid) {
  const char* src = (const char*)pb + (size_t)s * (STAGE * 8192);
  char* d = (char*)dst;
#pragma unroll
  for (int i = 0; i < (STAGE * 8192 / 16) / 256; ++i) {
    int o = (i * 256 + tid) * 16;
    ASYNC_COPY16(d + o, src + o);
  }
}

__global__ __launch_bounds__(256) void vq_argmin(const _Float16* __restrict__ pa,
                                                 const _Float16* __restrict__ pb,
                                                 const float* __restrict__ e2,
                                                 int* __restrict__ idxOut) {
  __shared__ _Float16 ldsB[2 * STAGE * 4096];  // 64 KB double buffer
  __shared__ float    ldsE2[VQ_K];             // 32 KB

  const int tid  = threadIdx.x;
  const int lane = tid & 31;
  const int wave = tid >> 5;

  // stage e2 fully in LDS
  {
    const v4f* s = (const v4f*)e2;
    v4f* d = (v4f*)ldsE2;
    for (int i = tid; i < VQ_K / 4; i += 256) d[i] = s[i];
  }

  // two resident A tiles (32 z-rows) per wave
  const int r0 = (blockIdx.x * 8 + wave) * 2;
  v16h a0[CH], a1[CH];
#pragma unroll
  for (int c = 0; c < CH; ++c) {
    a0[c] = *(const v16h*)(pa + ((size_t)(r0 * CH + c) * 32 + lane) * 16);
    a1[c] = *(const v16h*)(pa + ((size_t)((r0 + 1) * CH + c) * 32 + lane) * 16);
  }

  float best0[8], best1[8];
  int   bi0[8], bi1[8];
#pragma unroll
  for (int j = 0; j < 8; ++j) {
    best0[j] = -3.4e38f; best1[j] = -3.4e38f; bi0[j] = 0; bi1[j] = 0;
  }

  vq_fill_stage(pb, ldsB, 0, tid);
  for (int s = 0; s < NSTAGE; ++s) {
    vq_wait_async();
    __syncthreads();  // stage s visible to all waves
    if (s + 1 < NSTAGE)
      vq_fill_stage(pb, ldsB + ((s + 1) & 1) * (STAGE * 4096), s + 1, tid);
    const _Float16* buf = ldsB + (s & 1) * (STAGE * 4096);
    for (int kt = 0; kt < STAGE; ++kt) {
      v8f acc0 = {}, acc1 = {};
#pragma unroll
      for (int c = 0; c < CH; ++c) {
        v16h b = *(const v16h*)(buf + ((kt * CH + c) * 32 + lane) * 16);
        acc0 = __builtin_amdgcn_wmma_f32_16x16x32_f16(false, a0[c], false, b,
                                                      (short)0, acc0, false, false);
        acc1 = __builtin_amdgcn_wmma_f32_16x16x32_f16(false, a1[c], false, b,
                                                      (short)0, acc1, false, false);
      }
      int ktg = s * STAGE + kt;
      int col = ktg * 16 + (lane & 15);
      float e2v = ldsE2[col];
#pragma unroll
      for (int j = 0; j < 8; ++j) {
        float s0 = 2.0f * acc0[j] - e2v;   // maximize 2*dot - ||e||^2
        if (s0 > best0[j]) { best0[j] = s0; bi0[j] = col; }
        float s1 = 2.0f * acc1[j] - e2v;
        if (s1 > best1[j]) { best1[j] = s1; bi1[j] = col; }
      }
    }
    __syncthreads();  // all waves done reading this buffer before refill
  }

  // per-row argmax reduction across the 16-lane C-matrix groups.
  // C layout: VGPR j, lanes 0-15 -> row j ; lanes 16-31 -> row j+8.
#pragma unroll
  for (int j = 0; j < 8; ++j) {
    float v = best0[j]; int bi = bi0[j];
#pragma unroll
    for (int m = 1; m < 16; m <<= 1) {
      float ov = __shfl_xor(v, m, 32);
      int   oi = __shfl_xor(bi, m, 32);
      if (ov > v || (ov == v && oi < bi)) { v = ov; bi = oi; }
    }
    if ((lane & 15) == 0) idxOut[r0 * 16 + j + (lane >> 4) * 8] = bi;

    v = best1[j]; bi = bi1[j];
#pragma unroll
    for (int m = 1; m < 16; m <<= 1) {
      float ov = __shfl_xor(v, m, 32);
      int   oi = __shfl_xor(bi, m, 32);
      if (ov > v || (ov == v && oi < bi)) { v = ov; bi = oi; }
    }
    if ((lane & 15) == 0) idxOut[(r0 + 1) * 16 + j + (lane >> 4) * 8] = bi;
  }
}

// ---------------- gather outputs ----------------
__global__ __launch_bounds__(256) void vq_gather(const float* __restrict__ emb,
                                                 const int* __restrict__ idx,
                                                 float* __restrict__ out) {
  int row = blockIdx.x * 4 + (threadIdx.x >> 6);  // 4 rows per block
  int t   = threadIdx.x & 63;                     // 64 float4 per row
  int k   = idx[row];
  v4f v = ((const v4f*)(emb + (size_t)k * VQ_D))[t];
  v4f* q  = (v4f*)out;                                    // quantized
  v4f* st = (v4f*)(out + (size_t)VQ_N * VQ_D);            // straight_through
  q [(size_t)row * (VQ_D / 4) + t] = v;
  st[(size_t)row * (VQ_D / 4) + t] = v;
  if (t == 0) out[(size_t)2 * VQ_N * VQ_D + row] = (float)k;  // indices
}

extern "C" void kernel_launch(void* const* d_in, const int* in_sizes, int n_in,
                              void* d_out, int out_size, void* d_ws, size_t ws_size,
                              hipStream_t stream) {
  const float* z   = (const float*)d_in[0];  // [N, D] f32
  const float* emb = (const float*)d_in[1];  // [K, D] f32
  float* out = (float*)d_out;
  char* ws = (char*)d_ws;

  _Float16* pa = (_Float16*)ws;                                        // 32 MB
  _Float16* pb = (_Float16*)(ws + (size_t)VQ_N * VQ_D * 2);            //  4 MB
  float*    e2 = (float*)(ws + (size_t)VQ_N * VQ_D * 2 + (size_t)VQ_K * VQ_D * 2);
  int*     idx = (int*)((char*)e2 + (size_t)VQ_K * 4);                 // 256 KB

  vq_pack_z  <<<(NTILE * CH * 32) / 256, 256, 0, stream>>>(z, pa);
  vq_pack_emb<<<(KTILE * CH * 32) / 256, 256, 0, stream>>>(emb, pb);
  vq_rownorm <<<VQ_K / 256, 256, 0, stream>>>(emb, e2);
  vq_argmin  <<<NTILE / 16, 256, 0, stream>>>(pa, pb, e2, idx);
  vq_gather  <<<VQ_N / 4, 256, 0, stream>>>(emb, idx, out);
}
